// MultiHeadAttention_40690520162622
// MI455X (gfx1250) — matrix operand order repaired and checked by reading, not compiled, not run on hardware
//
#include <hip/hip_runtime.h>

// ---------------------------------------------------------------------------
// MHA forward for MI455X (gfx1250, wave32, WMMA + TDM).
//   B=2, S=2048, E=1024, H=16, D=64
// Pipeline: [weights f32->bf16] -> [QKV GEMMs (bf16 WMMA, f32 acc; Q
// pre-scaled by 1/sqrt(D)*log2(e))] -> [flash attention (bf16 WMMA, TDM
// double-buffered 64-key KV chunks, in-LDS V transpose, base-2 online
// softmax)] -> [out GEMM + bias]
// ---------------------------------------------------------------------------

#define S_LEN 2048
#define EMB   1024
#define NHEAD 16
#define HDIM  64
#define BATCH 2
#define MTOT  (BATCH * S_LEN)   // 4096
#define KCHUNK 64               // keys per attention chunk
#define CHUNK_ELEMS (KCHUNK * HDIM)   // 4096 bf16 = 8 KB

// 1/sqrt(HDIM) * log2(e): scores come out of WMMA already in log2 domain.
#define Q_SCALE 0.18033688011112042f

typedef __attribute__((ext_vector_type(16))) __bf16 v16bf;
typedef __attribute__((ext_vector_type(8)))  float  v8f;
typedef __attribute__((ext_vector_type(4)))  float  v4f;
typedef __attribute__((ext_vector_type(4)))  unsigned int v4u;
typedef __attribute__((ext_vector_type(4)))  int    i32x4;
typedef __attribute__((ext_vector_type(8)))  int    i32x8;

#if defined(__has_builtin)
#  if __has_builtin(__builtin_amdgcn_tensor_load_to_lds) && \
      __has_builtin(__builtin_amdgcn_s_wait_tensorcnt)
#    define HAVE_TDM 1
#  endif
#endif
#ifndef HAVE_TDM
#  define HAVE_TDM 0
#endif

union Frag16 {
    v16bf v;
    v4u   q[2];
    unsigned short s[16];
};

__device__ __forceinline__ unsigned short bf16b(float f) {
    union { float f; unsigned int u; } c;
    c.f = f;
    unsigned int u = c.u;
    u += 0x7FFFu + ((u >> 16) & 1u);   // round-to-nearest-even
    return (unsigned short)(u >> 16);
}

__device__ __forceinline__ v8f wmma_bf16(v16bf a, v16bf b, v8f c) {
    // D = A(16x32 bf16) * B(32x16 bf16) + C(16x16 f32)
    return __builtin_amdgcn_wmma_f32_16x16x32_bf16(
        false, a, false, b, (short)0, c, false, false);
}

// A fragment (16x32, bf16 source, row-major, ld elements).  ISA 7.12.2:
// lane: M = lane&15, kb = (lane>>4)*8; elems [0..7] = K kb.., [8..15] = K 16+kb..
__device__ __forceinline__ v16bf load_a_bf16(const unsigned short* base, int ld, int lane) {
    const int m  = lane & 15;
    const int kb = (lane >> 4) << 3;
    const unsigned short* p = base + (size_t)m * ld + kb;
    Frag16 f;
    f.q[0] = *(const v4u*)p;
    f.q[1] = *(const v4u*)(p + 16);
    return f.v;
}

// A fragment from f32 source, converted in-register to bf16.
__device__ __forceinline__ v16bf load_a_f32(const float* base, int ld, int lane) {
    const int m  = lane & 15;
    const int kb = (lane >> 4) << 3;
    const float* p0 = base + (size_t)m * ld + kb;
    const float* p1 = p0 + 16;
    v4f c0 = *(const v4f*)p0;
    v4f c1 = *(const v4f*)(p0 + 4);
    v4f c2 = *(const v4f*)p1;
    v4f c3 = *(const v4f*)(p1 + 4);
    Frag16 f;
#pragma unroll
    for (int i = 0; i < 4; ++i) {
        f.s[i]      = bf16b(c0[i]);
        f.s[4 + i]  = bf16b(c1[i]);
        f.s[8 + i]  = bf16b(c2[i]);
        f.s[12 + i] = bf16b(c3[i]);
    }
    return f.v;
}

// B fragment where B[k][n] = SRC[n][k] (row-major along k). Contiguous loads.
// lane: N = lane&15, kb = (lane>>4)*8; elems [0..7] = K kb.., [8..15] = K 16+kb..
__device__ __forceinline__ v16bf load_b_rowk(const unsigned short* base, int ld, int lane) {
    const int n  = lane & 15;
    const int kb = (lane >> 4) << 3;
    const unsigned short* p = base + (size_t)n * ld + kb;
    Frag16 f;
    f.q[0] = *(const v4u*)p;
    f.q[1] = *(const v4u*)(p + 16);
    return f.v;
}

__device__ __forceinline__ float rowmax16(float x) {
#pragma unroll
    for (int m = 1; m <= 8; m <<= 1) x = fmaxf(x, __shfl_xor(x, m, 32));
    return x;
}
__device__ __forceinline__ float rowsum16(float x) {
#pragma unroll
    for (int m = 1; m <= 8; m <<= 1) x += __shfl_xor(x, m, 32);
    return x;
}

// ---------------------------------------------------------------------------
// TDM: DMA one contiguous 8 KB chunk (4096 bf16) from global memory into LDS.
// D# packing per CDNA5 ISA 8.3/8.4 (group0: count/lds/global/type=2;
// group1: data_size=2B, tensor_dim0=tile_dim0=4096, dim0_stride=4096).
// ---------------------------------------------------------------------------
#if HAVE_TDM
typedef __attribute__((ext_vector_type(4))) unsigned int u32x4;
__device__ __forceinline__ void tdm_load_chunk(unsigned int lds_off,
                                               const unsigned short* gptr) {
    unsigned long long ga = (unsigned long long)(size_t)gptr;
    u32x4 g0;
    g0[0] = 1u;                                            // count=1, user D#
    g0[1] = lds_off;                                       // lds_addr (bytes)
    g0[2] = (unsigned int)(ga & 0xffffffffu);              // global_addr[31:0]
    g0[3] = (unsigned int)((ga >> 32) & 0x01ffffffu)       // global_addr[56:32]
            | (2u << 30);                                  // type = 2 (image)
    i32x8 g1;
    g1[0] = (int)(1u << 16);                  // data_size = 1 -> 2 bytes/elem
    g1[1] = (int)((CHUNK_ELEMS & 0xffffu) << 16);   // tensor_dim0[15:0]
    g1[2] = (int)(1u << 16);                  // tensor_dim0[31:16]=0, tensor_dim1=1
    g1[3] = (int)((CHUNK_ELEMS & 0xffffu) << 16);   // tile_dim0 = 4096
    g1[4] = 1;                                // tile_dim1 = 1, tile_dim2 = 0
    g1[5] = CHUNK_ELEMS;                      // tensor_dim0_stride[31:0]
    g1[6] = 0;
    g1[7] = 0;
    i32x4 z4 = (i32x4){0, 0, 0, 0};
#if __has_include(<hip/amd_detail/amd_gfx1250_TDM.h>)
    i32x8 z8 = (i32x8){0, 0, 0, 0, 0, 0, 0, 0};
    __builtin_amdgcn_tensor_load_to_lds(g0, g1, z4, z4, z8, 0);
#else
    __builtin_amdgcn_tensor_load_to_lds(g0, g1, z4, z4, 0);
#endif
}
#endif

// ---------------------------------------------------------------------------
// f32 -> bf16 elementwise conversion (weights)
// ---------------------------------------------------------------------------
__global__ void cvt_f32_bf16(const float* __restrict__ in,
                             unsigned short* __restrict__ out, int n) {
    int i = blockIdx.x * blockDim.x + threadIdx.x;
    if (i < n) out[i] = bf16b(in[i]);
}

// ---------------------------------------------------------------------------
// GEMM: C[M][N] = A[M][K] * W[N][K]^T    (torch linear: y = x @ W.T)
// MODE 0: A = f32, store bf16 into [B,H,S,D] layout (K/V projection)
// MODE 2: like MODE 0 but pre-scales by Q_SCALE (Q projection)
// MODE 1: A = bf16, store f32 + bias[n] into out_f (final projection)
// Block: 256 thr = 8 waves, block tile 128(M)x128(N), wave tile 32x64.
// ---------------------------------------------------------------------------
template <int MODE>
__global__ __launch_bounds__(256) void gemm_kernel(
    const float* __restrict__ Af,
    const unsigned short* __restrict__ Ab,
    const unsigned short* __restrict__ W,
    unsigned short* __restrict__ out_qkv,
    float* __restrict__ out_f,
    const float* __restrict__ bias,
    int M, int N, int K) {
    const int tid  = threadIdx.x;
    const int wave = tid >> 5;
    const int lane = tid & 31;
    const int m0 = blockIdx.y * 128 + (wave >> 1) * 32;
    const int n0 = blockIdx.x * 128 + (wave & 1) * 64;

    v8f acc[2][4];
#pragma unroll
    for (int i = 0; i < 2; ++i)
#pragma unroll
        for (int j = 0; j < 4; ++j) acc[i][j] = (v8f){};

    for (int k0 = 0; k0 < K; k0 += 32) {
        v16bf a0, a1;
        if (MODE == 1) {
            a0 = load_a_bf16(Ab + (size_t)m0 * K + k0, K, lane);
            a1 = load_a_bf16(Ab + (size_t)(m0 + 16) * K + k0, K, lane);
        } else {
            a0 = load_a_f32(Af + (size_t)m0 * K + k0, K, lane);
            a1 = load_a_f32(Af + (size_t)(m0 + 16) * K + k0, K, lane);
        }
#pragma unroll
        for (int j = 0; j < 4; ++j) {
            v16bf bw = load_b_rowk(W + (size_t)(n0 + j * 16) * K + k0, K, lane);
            acc[0][j] = wmma_bf16(a0, bw, acc[0][j]);
            acc[1][j] = wmma_bf16(a1, bw, acc[1][j]);
        }
    }

    const float sc = (MODE == 2) ? Q_SCALE : 1.0f;
    const int rbase = (lane >> 4) << 3;
    const int nl = lane & 15;
#pragma unroll
    for (int i = 0; i < 2; ++i) {
#pragma unroll
        for (int j = 0; j < 4; ++j) {
#pragma unroll
            for (int e = 0; e < 8; ++e) {
                const int m = m0 + i * 16 + rbase + e;
                const int n = n0 + j * 16 + nl;
                if (MODE == 1) {
                    out_f[(size_t)m * N + n] = acc[i][j][e] + bias[n];
                } else {
                    // [B,H,S,D]: b = m/S, s = m%S, h = n/D, d = n%D
                    const int bb = m >> 11, s = m & (S_LEN - 1);
                    const int h = n >> 6, d = n & (HDIM - 1);
                    out_qkv[((((size_t)bb * NHEAD + h) * S_LEN) + s) * HDIM + d] =
                        bf16b(acc[i][j][e] * sc);
                }
            }
        }
    }
}

// ---------------------------------------------------------------------------
// Flash attention: per (b,h), block = 8 waves x 16 q-rows = 128 q-rows.
// K/V chunks (64 keys) double-buffered in LDS via TDM; V transposed in LDS
// so every WMMA operand is a contiguous ds_load_b128 pair. Base-2 online
// softmax (Q pre-scaled by 1/sqrt(D)*log2(e)); one cross-lane reduction and
// one O-rescale per 64 keys. Two barriers per chunk.
// ---------------------------------------------------------------------------
__global__ __launch_bounds__(256) void attn_kernel(
    const unsigned short* __restrict__ Q,
    const unsigned short* __restrict__ K,
    const unsigned short* __restrict__ V,
    unsigned short* __restrict__ O) {
    __shared__ unsigned short lds_k[2][CHUNK_ELEMS];    // 2 x 8 KB, [64 key][64 d]
    __shared__ unsigned short lds_vs[2][CHUNK_ELEMS];   // 2 x 8 KB staging (row-major)
    __shared__ unsigned short lds_vt[2][CHUNK_ELEMS];   // 2 x 8 KB transposed [64 d][64 key]
    __shared__ unsigned short lds_p[8][16 * KCHUNK];    // 16 KB per-wave P bounce

    const int tid  = threadIdx.x;
    const int wave = tid >> 5;
    const int lane = tid & 31;
    const int bh = blockIdx.y;            // b*NHEAD + h
    const int b = bh >> 4, h = bh & 15;
    const int qbase = blockIdx.x * 128 + wave * 16;

    const unsigned short* Qh = Q + (size_t)bh * S_LEN * HDIM;
    const unsigned short* Kh = K + (size_t)bh * S_LEN * HDIM;
    const unsigned short* Vh = V + (size_t)bh * S_LEN * HDIM;

    // Q fragments are loop-invariant (16 rows x 64 d = two K-steps)
    const v16bf aq0 = load_a_bf16(Qh + (size_t)qbase * HDIM + 0, HDIM, lane);
    const v16bf aq1 = load_a_bf16(Qh + (size_t)qbase * HDIM + 32, HDIM, lane);

    v8f o0 = (v8f){}, o1 = (v8f){}, o2 = (v8f){}, o3 = (v8f){};
    float m_run[8], l_run[8];
#pragma unroll
    for (int e = 0; e < 8; ++e) { m_run[e] = -1e30f; l_run[e] = 0.0f; }

    const int rbase = (lane >> 4) << 3;
    const int nloc = lane & 15;
    unsigned short* pp = lds_p[wave];

#if HAVE_TDM
    const unsigned int koff[2] = {(unsigned int)(size_t)&lds_k[0][0],
                                  (unsigned int)(size_t)&lds_k[1][0]};
    const unsigned int voff[2] = {(unsigned int)(size_t)&lds_vs[0][0],
                                  (unsigned int)(size_t)&lds_vs[1][0]};
    // Prologue: DMA chunk 0 into buffer 0.
    if (wave == 0) {
        tdm_load_chunk(koff[0], Kh);
        tdm_load_chunk(voff[0], Vh);
    }
#endif

    for (int kc = 0; kc < S_LEN; kc += KCHUNK) {
        const int cur = (kc >> 6) & 1;

#if HAVE_TDM
        if (wave == 0) {
            if (kc + KCHUNK < S_LEN) {
                // Prefetch chunk kc+64 into the other buffer, then wait only
                // for the two TDM ops of the *current* chunk (in-order).
                tdm_load_chunk(koff[cur ^ 1], Kh + (size_t)(kc + KCHUNK) * HDIM);
                tdm_load_chunk(voff[cur ^ 1], Vh + (size_t)(kc + KCHUNK) * HDIM);
                __builtin_amdgcn_s_wait_tensorcnt((short)2);
            } else {
                __builtin_amdgcn_s_wait_tensorcnt((short)0);
            }
        }
#else
#pragma unroll
        for (int j = 0; j < 2; ++j) {
            ((v4u*)lds_k[cur])[tid + j * 256] =
                *(const v4u*)(Kh + (size_t)kc * HDIM + (size_t)(tid + j * 256) * 8);
            ((v4u*)lds_vs[cur])[tid + j * 256] =
                *(const v4u*)(Vh + (size_t)kc * HDIM + (size_t)(tid + j * 256) * 8);
        }
#endif
        __syncthreads();   // current chunk resident in LDS for all waves

        const unsigned short* ck = lds_k[cur];

        // Cooperative V transpose: vs[key][d] -> vt[d][key] (overlaps scores).
        {
            unsigned short* vt = lds_vt[cur];
#pragma unroll
            for (int j = 0; j < 2; ++j) {
                const int u = tid + j * 256;     // 0..511 8-elem units
                const int r = u >> 3;            // 0..63 (key)
                const int c0 = (u & 7) << 3;     // 0,8,..,56 (d)
                union { v4u q; unsigned short s[8]; } x;
                x.q = *(const v4u*)(lds_vs[cur] + (size_t)r * HDIM + c0);
#pragma unroll
                for (int i = 0; i < 8; ++i)
                    vt[(size_t)(c0 + i) * KCHUNK + r] = x.s[i];
            }
        }

        // Scores: S(16x64) = Q(16x64) . K_chunk(64x64)^T  -> 8 WMMAs
        v8f s[4];
#pragma unroll
        for (int t = 0; t < 4; ++t) {
            s[t] = (v8f){};
            s[t] = wmma_bf16(aq0, load_b_rowk(ck + (t * 16) * HDIM + 0, HDIM, lane), s[t]);
            s[t] = wmma_bf16(aq1, load_b_rowk(ck + (t * 16) * HDIM + 32, HDIM, lane), s[t]);
        }

        // Base-2 online softmax: rows live in VGPR index e (row = rbase+e),
        // cols across the 16 lanes of each half-wave. One cross-lane
        // reduction pass per 64 keys.
#pragma unroll
        for (int e = 0; e < 8; ++e) {
            const float x0 = s[0][e], x1 = s[1][e], x2 = s[2][e], x3 = s[3][e];
            float mx = rowmax16(fmaxf(fmaxf(x0, x1), fmaxf(x2, x3)));
            const float mnew = fmaxf(m_run[e], mx);
            const float p0 = exp2f(x0 - mnew);
            const float p1 = exp2f(x1 - mnew);
            const float p2 = exp2f(x2 - mnew);
            const float p3 = exp2f(x3 - mnew);
            const float rs = rowsum16((p0 + p1) + (p2 + p3));
            const float alpha = exp2f(m_run[e] - mnew);
            l_run[e] = l_run[e] * alpha + rs;
            m_run[e] = mnew;
            o0[e] *= alpha; o1[e] *= alpha; o2[e] *= alpha; o3[e] *= alpha;
            // Bounce P through LDS to convert C-layout -> A-layout
            unsigned short* pr = pp + (rbase + e) * KCHUNK + nloc;
            pr[0]  = bf16b(p0);
            pr[16] = bf16b(p1);
            pr[32] = bf16b(p2);
            pr[48] = bf16b(p3);
        }

        __syncthreads();   // vt[cur] fully written; everyone past K reads

        // O(16x64) += P(16x64) . V_chunk(64x64)  -> 8 WMMAs, all b128 loads.
        const unsigned short* vt = lds_vt[cur];
        const v16bf ap0 = load_a_bf16(pp + 0, KCHUNK, lane);
        const v16bf ap1 = load_a_bf16(pp + 32, KCHUNK, lane);
        o0 = wmma_bf16(ap0, load_b_rowk(vt + 0 * 16 * KCHUNK + 0, KCHUNK, lane), o0);
        o0 = wmma_bf16(ap1, load_b_rowk(vt + 0 * 16 * KCHUNK + 32, KCHUNK, lane), o0);
        o1 = wmma_bf16(ap0, load_b_rowk(vt + 1 * 16 * KCHUNK + 0, KCHUNK, lane), o1);
        o1 = wmma_bf16(ap1, load_b_rowk(vt + 1 * 16 * KCHUNK + 32, KCHUNK, lane), o1);
        o2 = wmma_bf16(ap0, load_b_rowk(vt + 2 * 16 * KCHUNK + 0, KCHUNK, lane), o2);
        o2 = wmma_bf16(ap1, load_b_rowk(vt + 2 * 16 * KCHUNK + 32, KCHUNK, lane), o2);
        o3 = wmma_bf16(ap0, load_b_rowk(vt + 3 * 16 * KCHUNK + 0, KCHUNK, lane), o3);
        o3 = wmma_bf16(ap1, load_b_rowk(vt + 3 * 16 * KCHUNK + 32, KCHUNK, lane), o3);
        // No trailing barrier: the next iteration's "resident" barrier plus
        // the vs/vt double-buffer split make the TDM prefetch targets safe.
    }

    // Finalize (divide by l) and store bf16 [B,S,E] with heads re-interleaved.
#pragma unroll
    for (int e = 0; e < 8; ++e) {
        const float inv = 1.0f / l_run[e];
        const int row = qbase + rbase + e;
        const size_t base = ((size_t)b * S_LEN + row) * EMB + (size_t)h * HDIM + nloc;
        O[base + 0]  = bf16b(o0[e] * inv);
        O[base + 16] = bf16b(o1[e] * inv);
        O[base + 32] = bf16b(o2[e] * inv);
        O[base + 48] = bf16b(o3[e] * inv);
    }
}

// ---------------------------------------------------------------------------
// Host launch
// ---------------------------------------------------------------------------
extern "C" void kernel_launch(void* const* d_in, const int* in_sizes, int n_in,
                              void* d_out, int out_size, void* d_ws, size_t ws_size,
                              hipStream_t stream) {
    const float* value = (const float*)d_in[0];
    const float* key   = (const float*)d_in[1];
    const float* query = (const float*)d_in[2];
    const float* Wv    = (const float*)d_in[3];
    const float* Wk    = (const float*)d_in[4];
    const float* Wq    = (const float*)d_in[5];
    const float* Wo    = (const float*)d_in[6];
    const float* bo    = (const float*)d_in[7];
    float* out = (float*)d_out;

    const size_t WE   = (size_t)EMB * EMB;                    // 1 Mi elems
    const size_t QKVE = (size_t)BATCH * NHEAD * S_LEN * HDIM; // 4 Mi elems

    unsigned short* ws = (unsigned short*)d_ws;
    unsigned short* wv_b = ws;
    unsigned short* wk_b = wv_b + WE;
    unsigned short* wq_b = wk_b + WE;
    unsigned short* wo_b = wq_b + WE;
    unsigned short* Qb   = wo_b + WE;
    unsigned short* Kb   = Qb + QKVE;
    unsigned short* Vb   = Kb + QKVE;
    unsigned short* AOb  = Vb + QKVE;   // attention output, [B,S,E] bf16

    // 1) Convert weights to bf16 once.
    const int cgrid = (int)(WE / 256);
    cvt_f32_bf16<<<cgrid, 256, 0, stream>>>(Wv, wv_b, (int)WE);
    cvt_f32_bf16<<<cgrid, 256, 0, stream>>>(Wk, wk_b, (int)WE);
    cvt_f32_bf16<<<cgrid, 256, 0, stream>>>(Wq, wq_b, (int)WE);
    cvt_f32_bf16<<<cgrid, 256, 0, stream>>>(Wo, wo_b, (int)WE);

    // 2) QKV projections (K/V plain, Q pre-scaled for base-2 softmax).
    dim3 ggrid(EMB / 128, MTOT / 128);   // (8, 32)
    gemm_kernel<0><<<ggrid, 256, 0, stream>>>(value, nullptr, wv_b, Vb, nullptr,
                                              nullptr, MTOT, EMB, EMB);
    gemm_kernel<0><<<ggrid, 256, 0, stream>>>(key, nullptr, wk_b, Kb, nullptr,
                                              nullptr, MTOT, EMB, EMB);
    gemm_kernel<2><<<ggrid, 256, 0, stream>>>(query, nullptr, wq_b, Qb, nullptr,
                                              nullptr, MTOT, EMB, EMB);

    // 3) Flash attention (TDM double-buffered 64-key KV chunks).
    dim3 agrid(S_LEN / 128, BATCH * NHEAD);   // (16, 32)
    attn_kernel<<<agrid, 256, 0, stream>>>(Qb, Kb, Vb, AOb);

    // 4) Output projection + bias (mode 1: bf16 A, f32 output).
    gemm_kernel<1><<<ggrid, 256, 0, stream>>>(nullptr, AOb, wo_b, nullptr, out,
                                              bo, MTOT, EMB, EMB);
}
